// DKVMN_78546361909953
// MI455X (gfx1250) — matrix operand order repaired and checked by reading, not compiled, not run on hardware
//
#include <hip/hip_runtime.h>
#include <math.h>

typedef __attribute__((ext_vector_type(16))) _Float16 v16h;
typedef __attribute__((ext_vector_type(8)))  _Float16 v8h;
typedef __attribute__((ext_vector_type(8)))  float    v8f;

#define WMMA_F16(a,b,c) __builtin_amdgcn_wmma_f32_16x16x32_f16(false,(a),false,(b),(short)0,(c),false,false)

#define LOG2E   1.4426950408889634f
#define LOG2E2  2.8853900817779268f

// Branch-free hardware transcendentals (v_tanh_f32 / v_exp_f32 / v_rcp_f32).
__device__ __forceinline__ float fast_exp2(float x) { return __builtin_amdgcn_exp2f(x); }
__device__ __forceinline__ float fast_rcp(float x)  { return __builtin_amdgcn_rcpf(x); }

__device__ __forceinline__ float fast_tanh(float x) {
#if __has_builtin(__builtin_amdgcn_tanhf)
    return __builtin_amdgcn_tanhf(x);
#else
    // tanh(x) = 1 - 2/(1 + exp2(2x*log2e)); saturates correctly at +-inf
    return 1.0f - 2.0f * fast_rcp(1.0f + fast_exp2(x * LOG2E2));
#endif
}

__device__ __forceinline__ float fast_sigmoid(float x) {
    return fast_rcp(1.0f + fast_exp2(-x * LOG2E));
}

// ---------------------------------------------------------------------------
// Load a 16x32 f16 A-fragment from LDS (row-major, rowStride f16 elements).
// A layout: lane = (row 0..15) + 16*half; slots 0..15 hold
// K = k0 + 16*half + {0..15} (matches swizzled B order).
// ---------------------------------------------------------------------------
__device__ __forceinline__ v16h ldsA(const _Float16* X, int rowStride, int lane, int k0) {
    const int row  = lane & 15;
    const int half = lane >> 4;
    const _Float16* p = X + row * rowStride + k0 + 16 * half;
    v8h lo = *(const v8h*)p;
    v8h hi = *(const v8h*)(p + 8);
    return __builtin_shufflevector(lo, hi, 0,1,2,3,4,5,6,7,8,9,10,11,12,13,14,15);
}

// ---------------------------------------------------------------------------
// Zero fill
// ---------------------------------------------------------------------------
__global__ void zero_kernel(float* __restrict__ p, int n) {
    int i = blockIdx.x * blockDim.x + threadIdx.x;
    if (i < n) p[i] = 0.0f;
}

// ---------------------------------------------------------------------------
// Swizzle a (K x 128) f32 weight into f16 WMMA B-fragment order:
// B order used here, matching ldsA (32x16 tile, f16): lane = 16*half + col; lanes 0-15 hold K=0..15,
// lanes 16-31 hold K=16..31; lane's 16 f16 slots are K_rel = 0..15 in order.
// dst[((kt*8 + nt)*32 + lane)*16 + slot]
// ---------------------------------------------------------------------------
__global__ void swz_kernel(const float* __restrict__ src, _Float16* __restrict__ dst, int K) {
    int i = blockIdx.x * blockDim.x + threadIdx.x;
    if (i >= K * 128) return;
    int k = i >> 7, n = i & 127;
    int kt = k >> 5, kk = k & 31, half = kk >> 4;
    int nt = n >> 4, col = n & 15;
    int lane = half * 16 + col;
    int slot = kk & 15;
    dst[((size_t)(kt * 8 + nt) * 32 + lane) * 16 + slot] = (_Float16)src[i];
}

// ---------------------------------------------------------------------------
// Attention: attn[b,s,:] = softmax(q_emb[b,s] @ key^T) over M=50.
// One 64-thread block per (b,s).
// ---------------------------------------------------------------------------
__global__ __launch_bounds__(64)
void attn_kernel(const int* __restrict__ qseq, const float* __restrict__ emb,
                 const float* __restrict__ key, float* __restrict__ attn) {
    constexpr int M = 50, D = 128;
    __shared__ __align__(16) float q[128];
    __shared__ float lg[64];
    __shared__ float red[2];
    const int blk = blockIdx.x;          // b*S + s
    const int t = threadIdx.x;
    const int qid = qseq[blk];
    q[t]      = emb[(size_t)qid * D + t];
    q[t + 64] = emb[(size_t)qid * D + t + 64];
    __syncthreads();
    if (t < M) {
        const float* kr = key + (size_t)t * D;
        float dot = 0.0f;
        for (int d = 0; d < D; d += 4) {
            float4 kv = *(const float4*)(kr + d);
            dot += q[d] * kv.x + q[d+1] * kv.y + q[d+2] * kv.z + q[d+3] * kv.w;
        }
        lg[t] = dot;
    }
    __syncthreads();
    if (t == 0) {
        float mx = lg[0];
        for (int m = 1; m < M; ++m) mx = fmaxf(mx, lg[m]);
        red[0] = mx;
    }
    __syncthreads();
    if (t < M) lg[t] = fast_exp2((lg[t] - red[0]) * LOG2E);
    __syncthreads();
    if (t == 0) {
        float sm = 0.0f;
        for (int m = 0; m < M; ++m) sm += lg[m];
        red[1] = fast_rcp(sm);
    }
    __syncthreads();
    if (t < M) attn[(size_t)blk * M + t] = lg[t] * red[1];
}

// ---------------------------------------------------------------------------
// Main sequential scan. 8 blocks x 256 threads (8 wave32s).
// Block bg handles batches [bg*16, bg*16+16). Wave w owns N-tile n0 = 16*w.
// Activations in LDS; v (B,M,D) L2-resident in global; swizzled f16 weight
// fragments get hoisted into VGPRs by the compiler (loop-invariant loads).
// ---------------------------------------------------------------------------
__global__ __launch_bounds__(256)
void dkvmn_scan(const int* __restrict__ qseq, const float* __restrict__ emb,
                const float* __restrict__ attn, float* __restrict__ v,
                const _Float16* __restrict__ w1z, const float* __restrict__ b1,
                const _Float16* __restrict__ w2z, const float* __restrict__ b2,
                const _Float16* __restrict__ erz, const float* __restrict__ erb,
                const _Float16* __restrict__ adz, const float* __restrict__ adb,
                const _Float16* __restrict__ ow1z, const float* __restrict__ ob1,
                const float* __restrict__ ow2, const float* __restrict__ ob2,
                float* __restrict__ out) {
    constexpr int S = 100, M = 50, D = 128;
    __shared__ __align__(16) _Float16 Xh[16 * 256];   // [read | qe] f16
    __shared__ __align__(16) _Float16 h16[16 * 128];
    __shared__ __align__(16) _Float16 up16[16 * 128];
    __shared__ float eS[16 * 128];
    __shared__ float aS[16 * 128];
    __shared__ float wS[16 * 64];                     // attn weights (padded)
    __shared__ float b1s[128], b2s[128], erbs[128], adbs[128], ob1s[128];
    __shared__ int qidx[16];

    const int tid  = threadIdx.x;
    const int bg   = blockIdx.x;
    const int lane = tid & 31;
    const int wv   = tid >> 5;        // wave id == N-tile index (8 tiles of 16)
    const int col  = lane & 15;
    const int half = lane >> 4;
    const int n0   = wv * 16;

    if (tid < 128) {
        b1s[tid]  = b1[tid];  b2s[tid]  = b2[tid];
        erbs[tid] = erb[tid]; adbs[tid] = adb[tid];
        ob1s[tid] = ob1[tid];
    }

    const int rb  = tid >> 4;          // local batch 0..15 (read/update role)
    const int rd0 = (tid & 15) * 8;    // 8 contiguous d per thread
    float* vb = v + ((size_t)(bg * 16 + rb)) * M * D;

    __syncthreads();

    for (int s = 0; s < S; ++s) {
        // ---- stage 1: attention weights + question indices -----------------
        if (tid < 16) qidx[tid] = qseq[(size_t)(bg * 16 + tid) * S + s];
        for (int i = tid; i < 16 * 64; i += 256) {
            int b = i >> 6, m = i & 63;
            wS[i] = (m < M) ? attn[((size_t)(bg * 16 + b) * S + s) * M + m] : 0.0f;
        }
        __syncthreads();

        // ---- stage 2: qe gather + memory read ------------------------------
        for (int i = tid; i < 16 * 128; i += 256) {
            int b = i >> 7, d = i & 127;
            Xh[b * 256 + 128 + d] = (_Float16)emb[(size_t)qidx[b] * D + d];
        }
        {
            float acc[8];
            #pragma unroll
            for (int j = 0; j < 8; ++j) acc[j] = 0.0f;
            const float* vp = vb + rd0;
            for (int m = 0; m < M; ++m) {
                float wm = wS[rb * 64 + m];
                float4 u0 = *(const float4*)(vp + m * D);
                float4 u1 = *(const float4*)(vp + m * D + 4);
                acc[0] += wm * u0.x; acc[1] += wm * u0.y;
                acc[2] += wm * u0.z; acc[3] += wm * u0.w;
                acc[4] += wm * u1.x; acc[5] += wm * u1.y;
                acc[6] += wm * u1.z; acc[7] += wm * u1.w;
            }
            #pragma unroll
            for (int j = 0; j < 8; ++j) Xh[rb * 256 + rd0 + j] = (_Float16)acc[j];
        }
        __syncthreads();

        // ---- stage 3: h = tanh(X @ W1 + b1)   (16x256 @ 256x128) -----------
        {
            float bias = b1s[n0 + col];
            v8f c = {bias, bias, bias, bias, bias, bias, bias, bias};
            #pragma unroll
            for (int kt = 0; kt < 8; ++kt) {
                v16h a = ldsA(Xh, 256, lane, kt * 32);
                v16h bf = ((const v16h*)w1z)[(size_t)(kt * 8 + wv) * 32 + lane];
                c = WMMA_F16(a, bf, c);
            }
            #pragma unroll
            for (int r = 0; r < 8; ++r)
                h16[(half * 8 + r) * 128 + n0 + col] = (_Float16)fast_tanh(c[r]);
        }
        __syncthreads();

        // ---- stage 4: up = h @ W2 + b2 -------------------------------------
        {
            float bias = b2s[n0 + col];
            v8f c = {bias, bias, bias, bias, bias, bias, bias, bias};
            #pragma unroll
            for (int kt = 0; kt < 4; ++kt) {
                v16h a = ldsA(h16, 128, lane, kt * 32);
                v16h bf = ((const v16h*)w2z)[(size_t)(kt * 8 + wv) * 32 + lane];
                c = WMMA_F16(a, bf, c);
            }
            #pragma unroll
            for (int r = 0; r < 8; ++r)
                up16[(half * 8 + r) * 128 + n0 + col] = (_Float16)c[r];
        }
        __syncthreads();

        // ---- stage 5: e = sigmoid(up@er+erb), a = tanh(up@ad+adb) ----------
        {
            v16h afr[4];
            #pragma unroll
            for (int kt = 0; kt < 4; ++kt) afr[kt] = ldsA(up16, 128, lane, kt * 32);
            float be = erbs[n0 + col], ba = adbs[n0 + col];
            v8f ce = {be, be, be, be, be, be, be, be};
            v8f ca = {ba, ba, ba, ba, ba, ba, ba, ba};
            #pragma unroll
            for (int kt = 0; kt < 4; ++kt)
                ce = WMMA_F16(afr[kt], ((const v16h*)erz)[(size_t)(kt * 8 + wv) * 32 + lane], ce);
            #pragma unroll
            for (int kt = 0; kt < 4; ++kt)
                ca = WMMA_F16(afr[kt], ((const v16h*)adz)[(size_t)(kt * 8 + wv) * 32 + lane], ca);
            #pragma unroll
            for (int r = 0; r < 8; ++r) {
                int o = (half * 8 + r) * 128 + n0 + col;
                eS[o] = fast_sigmoid(ce[r]);
                aS[o] = fast_tanh(ca[r]);
            }
        }
        __syncthreads();

        // ---- stage 6: v = v*(1 - w⊗e) + w⊗a --------------------------------
        {
            float ev[8], av[8];
            #pragma unroll
            for (int j = 0; j < 8; ++j) {
                ev[j] = eS[rb * 128 + rd0 + j];
                av[j] = aS[rb * 128 + rd0 + j];
            }
            float* vp = vb + rd0;
            for (int m = 0; m < M; ++m) {
                float wm = wS[rb * 64 + m];
                float4 u0 = *(const float4*)(vp + m * D);
                float4 u1 = *(const float4*)(vp + m * D + 4);
                u0.x = u0.x * (1.0f - wm * ev[0]) + wm * av[0];
                u0.y = u0.y * (1.0f - wm * ev[1]) + wm * av[1];
                u0.z = u0.z * (1.0f - wm * ev[2]) + wm * av[2];
                u0.w = u0.w * (1.0f - wm * ev[3]) + wm * av[3];
                u1.x = u1.x * (1.0f - wm * ev[4]) + wm * av[4];
                u1.y = u1.y * (1.0f - wm * ev[5]) + wm * av[5];
                u1.z = u1.z * (1.0f - wm * ev[6]) + wm * av[6];
                u1.w = u1.w * (1.0f - wm * ev[7]) + wm * av[7];
                *(float4*)(vp + m * D)     = u0;
                *(float4*)(vp + m * D + 4) = u1;
            }
        }
        __syncthreads();
    }

    // ---- output head: final read (w from s=99, updated v) ------------------
    {
        float acc[8];
        #pragma unroll
        for (int j = 0; j < 8; ++j) acc[j] = 0.0f;
        const float* vp = vb + rd0;
        for (int m = 0; m < M; ++m) {
            float wm = wS[rb * 64 + m];
            float4 u0 = *(const float4*)(vp + m * D);
            float4 u1 = *(const float4*)(vp + m * D + 4);
            acc[0] += wm * u0.x; acc[1] += wm * u0.y;
            acc[2] += wm * u0.z; acc[3] += wm * u0.w;
            acc[4] += wm * u1.x; acc[5] += wm * u1.y;
            acc[6] += wm * u1.z; acc[7] += wm * u1.w;
        }
        #pragma unroll
        for (int j = 0; j < 8; ++j) Xh[rb * 256 + rd0 + j] = (_Float16)acc[j];
    }
    __syncthreads();

    // h_out = relu(X @ out_w1 + out_b1) -> eS (reused as f32 buffer)
    {
        float bias = ob1s[n0 + col];
        v8f c = {bias, bias, bias, bias, bias, bias, bias, bias};
        #pragma unroll
        for (int kt = 0; kt < 8; ++kt) {
            v16h a = ldsA(Xh, 256, lane, kt * 32);
            v16h bf = ((const v16h*)ow1z)[(size_t)(kt * 8 + wv) * 32 + lane];
            c = WMMA_F16(a, bf, c);
        }
        #pragma unroll
        for (int r = 0; r < 8; ++r)
            eS[(half * 8 + r) * 128 + n0 + col] = fmaxf(c[r], 0.0f);
    }
    __syncthreads();

    if (tid < 16) {
        float acc = ob2[0];
        for (int d = 0; d < 128; ++d) acc += eS[tid * 128 + d] * ow2[d];
        out[bg * 16 + tid] = fast_sigmoid(acc);
    }
}

// ---------------------------------------------------------------------------
extern "C" void kernel_launch(void* const* d_in, const int* in_sizes, int n_in,
                              void* d_out, int out_size, void* d_ws, size_t ws_size,
                              hipStream_t stream) {
    (void)in_sizes; (void)n_in; (void)out_size; (void)ws_size;
    const int*   qseq = (const int*)d_in[0];
    // d_in[1] = answer_seq (unused by the reference)
    const float* emb  = (const float*)d_in[2];
    const float* key  = (const float*)d_in[3];
    const float* w1   = (const float*)d_in[4];
    const float* b1   = (const float*)d_in[5];
    const float* w2   = (const float*)d_in[6];
    const float* b2   = (const float*)d_in[7];
    const float* erw  = (const float*)d_in[8];
    const float* erb  = (const float*)d_in[9];
    const float* adw  = (const float*)d_in[10];
    const float* adb  = (const float*)d_in[11];
    const float* ow1  = (const float*)d_in[12];
    const float* ob1  = (const float*)d_in[13];
    const float* ow2  = (const float*)d_in[14];
    const float* ob2  = (const float*)d_in[15];
    float* out = (float*)d_out;

    char* ws = (char*)d_ws;
    float*    v    = (float*)ws;                   // 819200 f32  (3,276,800 B)
    float*    attn = (float*)(ws + 3276800);       // 640000 f32  (2,560,000 B)
    _Float16* w1z  = (_Float16*)(ws + 5836800);    // 32768 f16
    _Float16* w2z  = w1z + 32768;                  // 16384 f16
    _Float16* erz  = w2z + 16384;
    _Float16* adz  = erz + 16384;
    _Float16* ow1z = adz + 16384;                  // 32768 f16 (ends ~6.07 MB)

    zero_kernel<<<(819200 + 255) / 256, 256, 0, stream>>>(v, 819200);
    swz_kernel<<<(256 * 128 + 255) / 256, 256, 0, stream>>>(w1, w1z, 256);
    swz_kernel<<<(128 * 128 + 255) / 256, 256, 0, stream>>>(w2, w2z, 128);
    swz_kernel<<<(128 * 128 + 255) / 256, 256, 0, stream>>>(erw, erz, 128);
    swz_kernel<<<(128 * 128 + 255) / 256, 256, 0, stream>>>(adw, adz, 128);
    swz_kernel<<<(256 * 128 + 255) / 256, 256, 0, stream>>>(ow1, ow1z, 256);
    attn_kernel<<<128 * 100, 64, 0, stream>>>(qseq, emb, key, attn);
    dkvmn_scan<<<8, 256, 0, stream>>>(qseq, emb, attn, v,
                                      w1z, b1, w2z, b2, erz, erb, adz, adb,
                                      ow1z, ob1, ow2, ob2, out);
}
